// Self_Attn_63161789055310
// MI455X (gfx1250) — matrix operand order repaired and checked by reading, not compiled, not run on hardware
//
#include <hip/hip_runtime.h>
#include <hip/hip_bf16.h>

typedef _Float16 v16h __attribute__((ext_vector_type(16)));
typedef _Float16 v8h  __attribute__((ext_vector_type(8)));
typedef float    v8f  __attribute__((ext_vector_type(8)));

union V16U {
    v16h     v;
    v8h      h8[2];
    unsigned u[8];
};

static __device__ __forceinline__ unsigned pkh(float a, float b) {
    auto t = __builtin_amdgcn_cvt_pkrtz(a, b);      // v_cvt_pk_rtz_f16_f32
    return __builtin_bit_cast(unsigned, t);
}

// lane i <-> lane i^16 exchange without touching the LDS pipe:
// v_permlanex16_b32 with identity lane selects (pure VALU, no dscnt wait).
static __device__ __forceinline__ float xchg_half(float v) {
    unsigned u = __builtin_bit_cast(unsigned, v);
    unsigned r = __builtin_amdgcn_permlanex16(u, u, 0x76543210u, 0xfedcba98u, false, false);
    return __builtin_bit_cast(float, r);
}

#define WMMA_F16(A, B, C) \
    __builtin_amdgcn_wmma_f32_16x16x32_f16(false, (A), false, (B), (short)0, (C), false, false)

// ---------------------------------------------------------------------------
// Kernel 1: f/g/h = W{k,q,v} @ x + b  (tiny GEMM, fp32 VALU), emit f16 in
// WMMA-friendly layouts:
//   fT [B,N,E] row-major  (natural A-matrix rows, M=i, K=e)
//   gB [B,N/16,32 lanes,16 f16]  pre-swizzled B-matrix slots (K=e, N=j)
//   hT [B,E,N] row-major  (natural A-matrix rows, M=c, K=i)
// ---------------------------------------------------------------------------
__global__ __launch_bounds__(256, 1)
void proj_kernel(const float* __restrict__ x,
                 const float* __restrict__ Wk, const float* __restrict__ bk,
                 const float* __restrict__ Wq, const float* __restrict__ bq,
                 const float* __restrict__ Wv, const float* __restrict__ bv,
                 _Float16* __restrict__ fT, _Float16* __restrict__ gB,
                 _Float16* __restrict__ hT) {
    constexpr int C = 256, N = 4096, E = 32;
    const int t  = blockIdx.x * 256 + threadIdx.x;   // global pixel id
    const int b  = t >> 12;
    const int nn = t & (N - 1);
    const float* xb = x + ((size_t)b * C) * N + nn;

    float af[E], ag[E], ah[E];
#pragma unroll
    for (int e = 0; e < E; ++e) { af[e] = bk[e]; ag[e] = bq[e]; ah[e] = bv[e]; }

    for (int c0 = 0; c0 < C; c0 += 4) {
        const float xv0 = xb[(size_t)(c0 + 0) * N];
        const float xv1 = xb[(size_t)(c0 + 1) * N];
        const float xv2 = xb[(size_t)(c0 + 2) * N];
        const float xv3 = xb[(size_t)(c0 + 3) * N];
#pragma unroll
        for (int e = 0; e < E; ++e) {
            const float4 wk4 = *(const float4*)(Wk + (size_t)e * C + c0);
            af[e] = fmaf(wk4.x, xv0, fmaf(wk4.y, xv1, fmaf(wk4.z, xv2, fmaf(wk4.w, xv3, af[e]))));
            const float4 wq4 = *(const float4*)(Wq + (size_t)e * C + c0);
            ag[e] = fmaf(wq4.x, xv0, fmaf(wq4.y, xv1, fmaf(wq4.z, xv2, fmaf(wq4.w, xv3, ag[e]))));
            const float4 wv4 = *(const float4*)(Wv + (size_t)e * C + c0);
            ah[e] = fmaf(wv4.x, xv0, fmaf(wv4.y, xv1, fmaf(wv4.z, xv2, fmaf(wv4.w, xv3, ah[e]))));
        }
    }

    // fT: contiguous 64B row per pixel
    {
        unsigned q[16];
#pragma unroll
        for (int k = 0; k < 16; ++k) q[k] = pkh(af[2 * k], af[2 * k + 1]);
        uint4* dst = (uint4*)(fT + (size_t)t * E);
        dst[0] = uint4{q[0],  q[1],  q[2],  q[3]};
        dst[1] = uint4{q[4],  q[5],  q[6],  q[7]};
        dst[2] = uint4{q[8],  q[9],  q[10], q[11]};
        dst[3] = uint4{q[12], q[13], q[14], q[15]};
    }
    // gB: B-matrix slot layout: lane (nn&15) gets e=0..7,16..23, lane (nn&15)+16 gets e=8..15,24..31
    {
        const int jt = nn >> 4;
        _Float16* base = gB + (((size_t)b * (N / 16) + jt) * 32) * 16;
        unsigned q[16];
#pragma unroll
        for (int k = 0; k < 16; ++k) q[k] = pkh(ag[2 * k], ag[2 * k + 1]);
        uint4* d0 = (uint4*)(base + (size_t)(nn & 15) * 16);
        d0[0] = uint4{q[0], q[1], q[2], q[3]};
        d0[1] = uint4{q[8],  q[9],  q[10], q[11]};
        uint4* d1 = (uint4*)(base + (size_t)((nn & 15) + 16) * 16);
        d1[0] = uint4{q[4], q[5], q[6], q[7]};
        d1[1] = uint4{q[12], q[13], q[14], q[15]};
    }
    // hT: [E,N] rows; lane-coalesced b16 stores per e
#pragma unroll
    for (int e = 0; e < E; ++e)
        hT[((size_t)b * E + e) * N + nn] = (_Float16)ah[e];
}

// ---------------------------------------------------------------------------
// Kernel 2: fused flash attention, transposed formulation.
// One wave per 16-query tile. S'[i,j] = A_f x B_g ; softmax over i is per-lane.
// O'[c,j] += A_h x B_P, P staged through 1.25KB LDS slice.
// ---------------------------------------------------------------------------
__global__ __launch_bounds__(128, 1)
void attn_flash(const _Float16* __restrict__ fT,
                const _Float16* __restrict__ gB,
                const _Float16* __restrict__ hT,
                _Float16* __restrict__ vT) {
    constexpr int N = 4096, E = 32;
    constexpr int ROW = 80;                       // padded LDS row stride (bytes)
    constexpr float L2E = 1.4426950408889634f;

    __shared__ __align__(16) char lds_all[4 * 16 * ROW];
    const int wave  = threadIdx.x >> 5;
    const int lane  = threadIdx.x & 31;
    const int lhalf = lane >> 4;
    const int l16   = lane & 15;
    char* lds = lds_all + wave * 16 * ROW;

    const int tile = blockIdx.x * 4 + wave;       // 0..1023
    const int b    = tile >> 8;
    const int j0   = (tile & 255) << 4;

    // B matrix for scores: 16 query columns, loaded once (coalesced 32B/lane)
    V16U Bg;
    {
        const v8h* p = (const v8h*)(gB + (((size_t)b * (N / 16) + (j0 >> 4)) * 32 + lane) * 16);
        Bg.h8[0] = p[0];
        Bg.h8[1] = p[1];
    }

    v8f O0 = {}, O1 = {};
    float m = -3.0e38f, l = 0.0f;

    const _Float16* fbase = fT + (size_t)b * N * E;
    const _Float16* hbase = hT + (size_t)b * E * N;

    for (int i0 = 0; i0 < N; i0 += 32) {
        // prefetch next key/value block
        __builtin_prefetch(fbase + (size_t)(i0 + 32 + l16) * E, 0, 1);
        __builtin_prefetch(hbase + (size_t)l16 * N + (i0 + 32), 0, 1);

        // A tiles of f (rows = keys i0..i0+31)
        V16U Af0, Af1;
        {
            const char* r0 = (const char*)(fbase + (size_t)(i0 + l16) * E);
            Af0.h8[0] = *(const v8h*)(r0 + lhalf * 16);
            Af0.h8[1] = *(const v8h*)(r0 + 32 + lhalf * 16);
            const char* r1 = (const char*)(fbase + (size_t)(i0 + 16 + l16) * E);
            Af1.h8[0] = *(const v8h*)(r1 + lhalf * 16);
            Af1.h8[1] = *(const v8h*)(r1 + 32 + lhalf * 16);
        }
        v8f S0 = {}, S1 = {};
        S0 = WMMA_F16(Af0.v, Bg.v, S0);           // S'[i, j], K = e = 32 in one shot
        S1 = WMMA_F16(Af1.v, Bg.v, S1);

        // per-query max over this 32-key block (lane owns column j = l16)
        float tm = S0[0];
#pragma unroll
        for (int r = 1; r < 8; ++r) tm = fmaxf(tm, S0[r]);
#pragma unroll
        for (int r = 0; r < 8; ++r) tm = fmaxf(tm, S1[r]);
        tm = fmaxf(tm, xchg_half(tm));            // v_permlanex16 (no LDS)
        const float mn   = fmaxf(m, tm);
        const float corr = __builtin_amdgcn_exp2f((m - mn) * L2E);

        float ps = 0.0f;
        float P0[8], P1[8];
#pragma unroll
        for (int r = 0; r < 8; ++r) { P0[r] = __builtin_amdgcn_exp2f((S0[r] - mn) * L2E); ps += P0[r]; }
#pragma unroll
        for (int r = 0; r < 8; ++r) { P1[r] = __builtin_amdgcn_exp2f((S1[r] - mn) * L2E); ps += P1[r]; }
        ps += xchg_half(ps);                      // v_permlanex16 (no LDS)
        l = l * corr + ps;
        m = mn;
#pragma unroll
        for (int r = 0; r < 8; ++r) { O0[r] *= corr; O1[r] *= corr; }

        // stash P as f16 into LDS [j][i_local]  (i_local = 16*t + 8*lhalf + r)
        // (the 4 dword stores per group merge into ds_store_b128)
        {
            char* row = lds + l16 * ROW + lhalf * 16;
#pragma unroll
            for (int r = 0; r < 8; r += 2) {
                *(unsigned*)(row + r * 2)      = pkh(P0[r], P0[r + 1]);
                *(unsigned*)(row + 32 + r * 2) = pkh(P1[r], P1[r + 1]);
            }
        }
        asm volatile("s_wait_dscnt 0" ::: "memory");

        // reload P in B-matrix layout (lane = column j, K = 32 keys)
        V16U Bp;
        {
            const char* prow = lds + l16 * ROW + lhalf * 16;
            Bp.h8[0] = *(const v8h*)(prow);
            Bp.h8[1] = *(const v8h*)(prow + 32);
        }

        // A tiles of h (rows = channels c)
        V16U Ah0, Ah1;
        {
            const char* c0r = (const char*)(hbase + (size_t)l16 * N + i0);
            Ah0.h8[0] = *(const v8h*)(c0r + lhalf * 16);
            Ah0.h8[1] = *(const v8h*)(c0r + 32 + lhalf * 16);
            const char* c1r = (const char*)(hbase + (size_t)(16 + l16) * N + i0);
            Ah1.h8[0] = *(const v8h*)(c1r + lhalf * 16);
            Ah1.h8[1] = *(const v8h*)(c1r + 32 + lhalf * 16);
        }
        O0 = WMMA_F16(Ah0.v, Bp.v, O0);           // O'[c, j] accumulate
        O1 = WMMA_F16(Ah1.v, Bp.v, O1);
    }

    // normalize, write v^T as [B,N,E] f16 rows (A-matrix-ready for kernel 3)
    const float inv = 1.0f / l;
    char* row = (char*)(vT + ((size_t)b * N + j0 + l16) * E);
#pragma unroll
    for (int r = 0; r < 8; r += 2) {
        *(unsigned*)(row + (8 * lhalf + r) * 2)        = pkh(O0[r] * inv, O0[r + 1] * inv);
        *(unsigned*)(row + (16 + 8 * lhalf + r) * 2)   = pkh(O1[r] * inv, O1[r + 1] * inv);
    }
}

// ---------------------------------------------------------------------------
// Kernel 3: o^T[n,c] = v^T[n,:] x Wo[c,:]^T  (WMMA, K=E=32), fuse bias and
// y = gamma*o + x. One wave per 16-pixel tile, loop over 16 channel tiles.
// ---------------------------------------------------------------------------
__global__ __launch_bounds__(128, 1)
void out_proj(const _Float16* __restrict__ vT,
              const float* __restrict__ Wo, const float* __restrict__ bo,
              const float* __restrict__ x,  const float* __restrict__ gammaP,
              float* __restrict__ y, float* __restrict__ o) {
    constexpr int C = 256, N = 4096, E = 32;
    const int wave  = threadIdx.x >> 5;
    const int lane  = threadIdx.x & 31;
    const int lhalf = lane >> 4;
    const int l16   = lane & 15;
    const int tile  = blockIdx.x * 4 + wave;
    const int b     = tile >> 8;
    const int n0    = (tile & 255) << 4;
    const float gamma = *gammaP;

    V16U Av;                                       // A = v^T tile (M=n, K=e)
    {
        const char* vrow = (const char*)(vT + ((size_t)b * N + n0 + l16) * E);
        Av.h8[0] = *(const v8h*)(vrow + lhalf * 32);
        Av.h8[1] = *(const v8h*)(vrow + lhalf * 32 + 16);
    }

    for (int c0 = 0; c0 < C; c0 += 16) {
        const int c = c0 + l16;
        // B = Wo^T tile: lane = column c, K = e 0..15 / 16..31 (f32 -> f16)
        const float* wrow = Wo + (size_t)c * E + lhalf * 16;
        const float4 w0 = *(const float4*)(wrow);
        const float4 w1 = *(const float4*)(wrow + 4);
        const float4 w2 = *(const float4*)(wrow + 8);
        const float4 w3 = *(const float4*)(wrow + 12);
        V16U Bw;
        Bw.u[0] = pkh(w0.x, w0.y); Bw.u[1] = pkh(w0.z, w0.w);
        Bw.u[2] = pkh(w1.x, w1.y); Bw.u[3] = pkh(w1.z, w1.w);
        Bw.u[4] = pkh(w2.x, w2.y); Bw.u[5] = pkh(w2.z, w2.w);
        Bw.u[6] = pkh(w3.x, w3.y); Bw.u[7] = pkh(w3.z, w3.w);

        v8f acc = {};
        acc = WMMA_F16(Av.v, Bw.v, acc);          // D[n, c]

        const float bias = bo[c];
        const size_t base = ((size_t)b * C + c) * N + n0 + 8 * lhalf;
        const float4 xa = *(const float4*)(x + base);
        const float4 xb4 = *(const float4*)(x + base + 4);

        float ov[8];
#pragma unroll
        for (int r = 0; r < 8; ++r) ov[r] = acc[r] + bias;

        *(float4*)(o + base)     = float4{ov[0], ov[1], ov[2], ov[3]};
        *(float4*)(o + base + 4) = float4{ov[4], ov[5], ov[6], ov[7]};
        *(float4*)(y + base)     = float4{fmaf(gamma, ov[0], xa.x),  fmaf(gamma, ov[1], xa.y),
                                          fmaf(gamma, ov[2], xa.z),  fmaf(gamma, ov[3], xa.w)};
        *(float4*)(y + base + 4) = float4{fmaf(gamma, ov[4], xb4.x), fmaf(gamma, ov[5], xb4.y),
                                          fmaf(gamma, ov[6], xb4.z), fmaf(gamma, ov[7], xb4.w)};
    }
}

// ---------------------------------------------------------------------------
extern "C" void kernel_launch(void* const* d_in, const int* in_sizes, int n_in,
                              void* d_out, int out_size, void* d_ws, size_t ws_size,
                              hipStream_t stream) {
    (void)in_sizes; (void)n_in; (void)out_size; (void)ws_size;
    const float* x     = (const float*)d_in[0];
    const float* Wk    = (const float*)d_in[1];
    const float* bk    = (const float*)d_in[2];
    const float* Wq    = (const float*)d_in[3];
    const float* bq    = (const float*)d_in[4];
    const float* Wv    = (const float*)d_in[5];
    const float* bv    = (const float*)d_in[6];
    const float* Wo    = (const float*)d_in[7];
    const float* bo    = (const float*)d_in[8];
    const float* gamma = (const float*)d_in[9];

    float* y = (float*)d_out;
    float* o = y + (size_t)4 * 256 * 4096;

    char* ws = (char*)d_ws;                        // 4 x 1MB f16 buffers
    _Float16* fT = (_Float16*)(ws);
    _Float16* gB = (_Float16*)(ws + (1u << 20));
    _Float16* hT = (_Float16*)(ws + (2u << 20));
    _Float16* vT = (_Float16*)(ws + (3u << 20));

    proj_kernel<<<64, 256, 0, stream>>>(x, Wk, bk, Wq, bq, Wv, bv, fT, gB, hT);
    attn_flash<<<256, 128, 0, stream>>>(fT, gB, hT, vT);
    out_proj<<<256, 128, 0, stream>>>(vT, Wo, bo, x, gamma, y, o);
}
